// GPT2Attention_27075473834130
// MI455X (gfx1250) — compile-verified, matrix operand order checked by reading, not run on hardware
//
#include <hip/hip_runtime.h>

// ---------------------------------------------------------------------------
// GPT-2 attention forward for MI455X (gfx1250, wave32, WMMA).
// B=2, T=2048, C=1024, H=16, D=64.  All matmuls via v_wmma_f32_16x16x32_f16.
// Workspace (f16): q,k,v [B,H,T,D] (8MB each) + attn out [B,T,C] (8MB) = 32MB.
// ---------------------------------------------------------------------------

typedef _Float16 v16h __attribute__((ext_vector_type(16)));
typedef _Float16 h8   __attribute__((ext_vector_type(8)));
typedef float    v8f  __attribute__((ext_vector_type(8)));

union F16Frag { v16h v; h8 h[2]; };

#define BM 128
#define BN 128
#define BK 32
#define LSTRIDE 40   // 32 + 8 halfs pad; 80B rows keep 16B alignment, spread banks

// --- gfx1250 async copy: global -> LDS, 16 bytes per active lane ------------
// ISA 15.18.3 opcode 98 (GLOBAL_LOAD_ASYNC_TO_LDS_B128), GV addressing.
// lds_off = AS(3) byte offset (low 32 bits of generic shared pointer).
__device__ inline void async_copy_b128_to_lds(unsigned int lds_off, const void* gptr) {
    asm volatile("global_load_async_to_lds_b128 %0, %1, off"
                 :: "v"(lds_off), "v"((unsigned long long)(uintptr_t)gptr)
                 : "memory");
}
__device__ inline void wait_asynccnt0() {
    asm volatile("s_wait_asynccnt 0x0" ::: "memory");
}

// ---------------------------------------------------------------------------
// Generic GEMM: C[M,N] = A[M,K] (f32 or f16) * B[K,N] (f32) + bias
// OUTMODE 0: f32 row-major to Cout.  OUTMODE 1: scatter f16 into Q/K/V [B,H,T,D].
// Block: 256 thr = 8 waves in 4x2; wave does 32x64 = 2x4 wmma tiles.
// ---------------------------------------------------------------------------
template <typename TA, int OUTMODE>
__global__ __launch_bounds__(256) void wmma_gemm_kernel(
    const TA* __restrict__ A, const float* __restrict__ Bm,
    const float* __restrict__ bias, float* __restrict__ Cout,
    _Float16* __restrict__ Qo, _Float16* __restrict__ Ko, _Float16* __restrict__ Vo,
    int M, int N, int K)
{
    __shared__ alignas(16) _Float16 Ah[BM][LSTRIDE];
    __shared__ alignas(16) _Float16 Bt[BN][LSTRIDE];   // [n][k] (transposed)

    const int tid  = threadIdx.x;
    const int lane = tid & 31;
    const int w    = tid >> 5;     // 0..7
    const int wr   = w >> 1;       // 0..3 row waves (32 rows each)
    const int wc   = w & 1;        // 0..1 col waves (64 cols each)
    const int lm   = lane & 15;
    const int lg   = lane >> 4;
    const int m0   = blockIdx.y * BM;
    const int n0   = blockIdx.x * BN;

    v8f acc[2][4] = {{{}, {}, {}, {}}, {{}, {}, {}, {}}};

    const int kTiles = K / BK;
    for (int kt = 0; kt < kTiles; ++kt) {
        const int kk = kt * BK;
        __syncthreads();
        // stage A tile (128x32) -> f16 LDS. 4096 elems / 256 thr = 16 each.
        #pragma unroll
        for (int i = 0; i < 16; ++i) {
            int l = tid + 256 * i;
            int r = l >> 5, c = l & 31;
            Ah[r][c] = (_Float16)(float)A[(size_t)(m0 + r) * K + (kk + c)];
        }
        // stage B tile transposed (128n x 32k). 4096 elems / 256 = 16 each.
        #pragma unroll
        for (int i = 0; i < 16; ++i) {
            int l = tid + 256 * i;
            int kr = l >> 7, nc = l & 127;
            Bt[nc][kr] = (_Float16)Bm[(size_t)(kk + kr) * N + (n0 + nc)];
        }
        if (kt + 1 < kTiles) {  // hint next weight tile toward L2 (global_prefetch_b8)
            __builtin_prefetch(&Bm[(size_t)(kk + BK) * N + n0 + (tid & 127)], 0, 1);
            __builtin_prefetch(&A[(size_t)(m0 + (tid >> 1)) * K + kk + BK], 0, 1);
        }
        __syncthreads();

        // fragments: k = e + 16*lg (same bijection for A and B -> sum-over-K exact)
        F16Frag afr[2], bfr[4];
        #pragma unroll
        for (int ti = 0; ti < 2; ++ti) {
            const _Float16* p = &Ah[wr * 32 + ti * 16 + lm][16 * lg];
            afr[ti].h[0] = *(const h8*)p;
            afr[ti].h[1] = *(const h8*)(p + 8);
        }
        #pragma unroll
        for (int tj = 0; tj < 4; ++tj) {
            const _Float16* p = &Bt[wc * 64 + tj * 16 + lm][16 * lg];
            bfr[tj].h[0] = *(const h8*)p;
            bfr[tj].h[1] = *(const h8*)(p + 8);
        }
        #pragma unroll
        for (int ti = 0; ti < 2; ++ti)
            #pragma unroll
            for (int tj = 0; tj < 4; ++tj)
                acc[ti][tj] = __builtin_amdgcn_wmma_f32_16x16x32_f16(
                    false, afr[ti].v, false, bfr[tj].v,
                    (short)0, acc[ti][tj], false, false);
    }

    // epilogue. C/D layout: vgpr r -> row m = r + 8*lg, col n = lm.
    #pragma unroll
    for (int ti = 0; ti < 2; ++ti)
        #pragma unroll
        for (int tj = 0; tj < 4; ++tj)
            #pragma unroll
            for (int r = 0; r < 8; ++r) {
                int m = m0 + wr * 32 + ti * 16 + r + 8 * lg;
                int n = n0 + wc * 64 + tj * 16 + lm;
                float val = acc[ti][tj][r] + bias[n];
                if (OUTMODE == 0) {
                    Cout[(size_t)m * N + n] = val;
                } else {
                    int which = n >> 10, cc = n & 1023;
                    int hh = cc >> 6, dd = cc & 63;
                    int bb = m >> 11, tt = m & 2047;
                    size_t idx = (((size_t)(bb * 16 + hh) * 2048) + tt) * 64 + dd;
                    _Float16* dst = (which == 0) ? Qo : (which == 1) ? Ko : Vo;
                    dst[idx] = (_Float16)val;
                }
            }
}

// ---------------------------------------------------------------------------
// Flash attention (causal). Block = 128 q rows (8 waves x 16 rows), one (b,h).
// Streams 32-key tiles: K [key][d] via async global->LDS copy, V transposed
// [d][key] synchronously (transpose can't be expressed as a byte copy).
// ---------------------------------------------------------------------------
#define QT 128

__global__ __launch_bounds__(256) void flash_attn_kernel(
    const _Float16* __restrict__ Q, const _Float16* __restrict__ Kd,
    const _Float16* __restrict__ Vd, _Float16* __restrict__ O /* [B,T,C] */)
{
    __shared__ alignas(16) _Float16 Klds[32][64];        // [key][d]
    __shared__ alignas(16) _Float16 Vt[64][LSTRIDE];     // [d][key]
    __shared__ alignas(16) _Float16 Plds[8][16][LSTRIDE];// per-wave P staging

    const int tid  = threadIdx.x;
    const int lane = tid & 31;
    const int w    = tid >> 5;
    const int lm   = lane & 15;
    const int lg   = lane >> 4;
    const int bh   = blockIdx.y;             // 0..31 = b*16+h
    const int qb   = blockIdx.x * QT;
    const int qw   = qb + w * 16;            // this wave's first q row

    const _Float16* Qh = Q  + (size_t)bh * 2048 * 64;
    const _Float16* Kh = Kd + (size_t)bh * 2048 * 64;
    const _Float16* Vh = Vd + (size_t)bh * 2048 * 64;

    // LDS byte offset of this thread's K-tile destination (16B chunk per lane)
    const int l0 = tid * 8;                              // half index in 32x64 tile
    const unsigned int klds_off =
        (unsigned int)(uintptr_t)(&Klds[0][0]) + (unsigned int)(l0 * 2);

    // Q fragments for d in [0,32) and [32,64), pre-scaled by 1/sqrt(D)=0.125
    F16Frag aq[2];
    {
        const _Float16* qp = Qh + (size_t)(qw + lm) * 64 + 16 * lg;
        #pragma unroll
        for (int d0 = 0; d0 < 2; ++d0) {
            h8 x0 = *(const h8*)(qp + d0 * 32);
            h8 x1 = *(const h8*)(qp + d0 * 32 + 8);
            #pragma unroll
            for (int e = 0; e < 8; ++e) {
                x0[e] = x0[e] * (_Float16)0.125f;
                x1[e] = x1[e] * (_Float16)0.125f;
            }
            aq[d0].h[0] = x0; aq[d0].h[1] = x1;
        }
    }

    float mrun[8], lrun[8];
    v8f oacc[4] = {{}, {}, {}, {}};
    #pragma unroll
    for (int r = 0; r < 8; ++r) { mrun[r] = -1e30f; lrun[r] = 0.f; }

    const int nkt = (qb + QT) / 32;          // key tiles of 32 up to causal edge
    for (int kt = 0; kt < nkt; ++kt) {
        const int kbase = kt * 32;
        __syncthreads();                     // previous tile fully consumed
        // stage K tile (32x64 f16 = 4KB): one async 16B copy per lane
        {
            int key = l0 >> 6, dd = l0 & 63;
            async_copy_b128_to_lds(klds_off, Kh + (size_t)(kbase + key) * 64 + dd);
        }
        // stage V tile transposed: h8 coalesced read, b16 scatter into Vt
        {
            int key = l0 >> 6, dd0 = l0 & 63;
            h8 vv = *(const h8*)(Vh + (size_t)(kbase + key) * 64 + dd0);
            #pragma unroll
            for (int e = 0; e < 8; ++e) Vt[dd0 + e][key] = vv[e];
        }
        wait_asynccnt0();                    // K-tile bytes landed in LDS
        __syncthreads();

        if (kbase <= qw + 15) {              // wave has unmasked work in this tile
            // K^T fragments: n = key, k = d
            F16Frag bk[2][2];
            #pragma unroll
            for (int j = 0; j < 2; ++j) {
                const _Float16* kp = &Klds[16 * j + lm][16 * lg];
                bk[j][0].h[0] = *(const h8*)kp;
                bk[j][0].h[1] = *(const h8*)(kp + 8);
                bk[j][1].h[0] = *(const h8*)(kp + 32);
                bk[j][1].h[1] = *(const h8*)(kp + 40);
            }
            v8f s[2] = {{}, {}};
            #pragma unroll
            for (int j = 0; j < 2; ++j) {
                s[j] = __builtin_amdgcn_wmma_f32_16x16x32_f16(
                    false, aq[0].v, false, bk[j][0].v, (short)0, s[j], false, false);
                s[j] = __builtin_amdgcn_wmma_f32_16x16x32_f16(
                    false, aq[1].v, false, bk[j][1].v, (short)0, s[j], false, false);
            }
            // causal mask + row max (rows live across 16-lane half-waves)
            float rowmax[8];
            #pragma unroll
            for (int r = 0; r < 8; ++r) {
                int mabs = qw + r + 8 * lg;
                #pragma unroll
                for (int j = 0; j < 2; ++j) {
                    int nabs = kbase + 16 * j + lm;
                    if (nabs > mabs) s[j][r] = -1e30f;
                }
                rowmax[r] = fmaxf(s[0][r], s[1][r]);
            }
            #pragma unroll
            for (int off = 8; off > 0; off >>= 1)
                #pragma unroll
                for (int r = 0; r < 8; ++r)
                    rowmax[r] = fmaxf(rowmax[r], __shfl_xor(rowmax[r], off, 32));

            float alpha[8], psum[8];
            v8f p0 = {}, p1 = {};
            #pragma unroll
            for (int r = 0; r < 8; ++r) {
                float nm = fmaxf(mrun[r], rowmax[r]);
                alpha[r] = __expf(mrun[r] - nm);
                mrun[r] = nm;
                p0[r] = __expf(s[0][r] - nm);
                p1[r] = __expf(s[1][r] - nm);
                psum[r] = p0[r] + p1[r];
            }
            #pragma unroll
            for (int off = 8; off > 0; off >>= 1)
                #pragma unroll
                for (int r = 0; r < 8; ++r)
                    psum[r] += __shfl_xor(psum[r], off, 32);
            #pragma unroll
            for (int r = 0; r < 8; ++r) lrun[r] = alpha[r] * lrun[r] + psum[r];
            #pragma unroll
            for (int dt = 0; dt < 4; ++dt)
                #pragma unroll
                for (int r = 0; r < 8; ++r) oacc[dt][r] *= alpha[r];

            // D-layout -> A-fragment layout via per-wave LDS patch (in-wave,
            // DS ops in-order; compiler inserts s_wait_dscnt)
            #pragma unroll
            for (int r = 0; r < 8; ++r) {
                int row = r + 8 * lg;
                Plds[w][row][lm]      = (_Float16)p0[r];
                Plds[w][row][16 + lm] = (_Float16)p1[r];
            }
            F16Frag ap;
            const _Float16* pp = &Plds[w][lm][16 * lg];
            ap.h[0] = *(const h8*)pp;
            ap.h[1] = *(const h8*)(pp + 8);

            // O += P * V  (contraction over 32 keys)
            #pragma unroll
            for (int dt = 0; dt < 4; ++dt) {
                F16Frag bv;
                const _Float16* vp = &Vt[16 * dt + lm][16 * lg];
                bv.h[0] = *(const h8*)vp;
                bv.h[1] = *(const h8*)(vp + 8);
                oacc[dt] = __builtin_amdgcn_wmma_f32_16x16x32_f16(
                    false, ap.v, false, bv.v, (short)0, oacc[dt], false, false);
            }
        }
    }

    // epilogue: write [B,T,C] f16 (input to projection GEMM)
    const int bb = bh >> 4, hh = bh & 15;
    #pragma unroll
    for (int dt = 0; dt < 4; ++dt)
        #pragma unroll
        for (int r = 0; r < 8; ++r) {
            int tt = qw + r + 8 * lg;
            int col = hh * 64 + dt * 16 + lm;
            float val = oacc[dt][r] / lrun[r];
            O[((size_t)(bb * 2048 + tt)) * 1024 + col] = (_Float16)val;
        }
}

// ---------------------------------------------------------------------------
extern "C" void kernel_launch(void* const* d_in, const int* in_sizes, int n_in,
                              void* d_out, int out_size, void* d_ws, size_t ws_size,
                              hipStream_t stream) {
    const float* x      = (const float*)d_in[0];  // [2,2048,1024]
    const float* W_attn = (const float*)d_in[1];  // [1024,3072]
    const float* b_attn = (const float*)d_in[2];  // [3072]
    const float* W_proj = (const float*)d_in[3];  // [1024,1024]
    const float* b_proj = (const float*)d_in[4];  // [1024]
    float* out = (float*)d_out;                   // [2,2048,1024] f32

    const size_t HE = (size_t)2 * 16 * 2048 * 64; // 4 Mi elems per q/k/v
    _Float16* q    = (_Float16*)d_ws;
    _Float16* k    = q + HE;
    _Float16* v    = k + HE;
    _Float16* attn = v + HE;                      // [B,T,C], total ws use = 32 MB

    dim3 blk(256);
    // 1) QKV GEMM: [4096,1024] x [1024,3072], scatter f16 q/k/v in [B,H,T,D]
    wmma_gemm_kernel<float, 1><<<dim3(3072 / BN, 4096 / BM), blk, 0, stream>>>(
        x, W_attn, b_attn, nullptr, q, k, v, 4096, 3072, 1024);
    // 2) causal flash attention per (b,h), 128 q rows per block
    flash_attn_kernel<<<dim3(2048 / QT, 32), blk, 0, stream>>>(q, k, v, attn);
    // 3) projection GEMM: [4096,1024] x [1024,1024] -> f32 output
    wmma_gemm_kernel<_Float16, 0><<<dim3(1024 / BN, 4096 / BM), blk, 0, stream>>>(
        attn, W_proj, b_proj, out, nullptr, nullptr, nullptr, 4096, 1024, 1024);
}
